// PSFFN_62723702391246
// MI455X (gfx1250) — matrix operand order repaired and checked by reading, not compiled, not run on hardware
//
#include <hip/hip_runtime.h>
#include <cmath>

typedef __attribute__((ext_vector_type(16))) _Float16 v16h;
typedef __attribute__((ext_vector_type(8)))  _Float16 v8h;
typedef __attribute__((ext_vector_type(8)))  float    v8f;

#define NB    4
#define NDIM  64
#define NCH   170
#define NCH2  340
#define NCH2P 352     // padded M for K1 (22 tiles of 16) -> unconditional stores
#define HWDIM 256
#define NPIX  65536   // 256*256
#define PP    8
#define PFP   5

__device__ __forceinline__ v16h cat8(v8h lo, v8h hi) {
    return __builtin_shufflevector(lo, hi, 0, 1, 2, 3, 4, 5, 6, 7,
                                           8, 9, 10, 11, 12, 13, 14, 15);
}

// ---------------------------------------------------------------------------
// K1: in-projection GEMM  x_in[b, m, n] = sum_k w_in[m,k] * x[b,k,n]
//     M = 340 (pad 352, 22 M-tiles), K = 64 (2 k-steps of 32), N tile = 128.
//     Operands staged in LDS as f16 (zero pad applied at staging); x_in is
//     padded to 352 rows so the store epilogue is branch-free.
// ---------------------------------------------------------------------------
__global__ __launch_bounds__(256)
void psffn_k1_inproj(const float* __restrict__ x, const float* __restrict__ w_in,
                     _Float16* __restrict__ xin)
{
    __shared__ __align__(16) _Float16 sw[NCH2P * 64]; // [m][k], rows >= 340 zeroed
    __shared__ __align__(16) _Float16 sx[128][72];    // [n][k], pad 64->72
    const int b  = blockIdx.y;
    const int n0 = blockIdx.x * 128;
    const int t  = threadIdx.x;

    for (int i = t; i < NCH2P * 64; i += 256) {
        const int m = i >> 6, k = i & 63;
        sw[i] = (m < NCH2) ? (_Float16)w_in[m * NDIM + k] : (_Float16)0.f;
    }
    for (int i = t; i < 64 * 128; i += 256) {
        const int k = i >> 7, n = i & 127;
        sx[n][k] = (_Float16)x[((size_t)b * NDIM + k) * NPIX + n0 + n];
    }
    __syncthreads();

    const int wave = t >> 5, lane = t & 31;
    const int nsub = lane & 15, hh = lane >> 4;
    const v8f vzero = {0.f, 0.f, 0.f, 0.f, 0.f, 0.f, 0.f, 0.f};

    for (int mt = wave; mt < 22; mt += 8) {           // 22 M-tiles over 8 waves
        v8f acc[8];
        #pragma unroll
        for (int i = 0; i < 8; ++i) acc[i] = vzero;

        #pragma unroll
        for (int ks = 0; ks < 2; ++ks) {
            // A fragment: ISA 16-bit A 16x32 layout -> two contiguous 8-half K runs
            const _Float16* pa = &sw[(mt * 16 + nsub) * 64 + ks * 32 + hh * 8];
            const v16h a = cat8(*(const v8h*)pa, *(const v8h*)(pa + 16));
            #pragma unroll
            for (int nt = 0; nt < 8; ++nt) {
                // B fragment: same K order as A -> two 8-half K runs (hh*8, 16+hh*8)
                const _Float16* pb = &sx[nt * 16 + nsub][ks * 32 + hh * 8];
                const v16h bf = cat8(*(const v8h*)pb, *(const v8h*)(pb + 16));
                acc[nt] = __builtin_amdgcn_wmma_f32_16x16x32_f16(
                              false, a, false, bf, (short)0, acc[nt], false, false);
            }
        }
        // D layout: VGPR v -> M = v + 8*hh, col = lane%16. Padded dst: no guards.
        #pragma unroll
        for (int nt = 0; nt < 8; ++nt) {
            #pragma unroll
            for (int v = 0; v < 8; ++v) {
                const int m = mt * 16 + v + 8 * hh;
                xin[((size_t)b * NCH2P + m) * NPIX + n0 + nt * 16 + nsub] =
                    (_Float16)acc[nt][v];
            }
        }
    }
}

// ---------------------------------------------------------------------------
// K2: depthwise 3x3 conv on x_in, exact-GELU gate, feat = gelu(x1)*x2,
//     plus per-tile partial sums for the global average pool.
// ---------------------------------------------------------------------------
__global__ __launch_bounds__(256)
void psffn_k2_dwgate(const _Float16* __restrict__ xin, const float* __restrict__ w_dw,
                     _Float16* __restrict__ feat, float* __restrict__ psum)
{
    __shared__ float s1[34][36];
    __shared__ float s2[34][36];
    __shared__ float red[256];
    const int b = blockIdx.z, c = blockIdx.y, tile = blockIdx.x;
    const int th0 = (tile >> 3) * 32, tw0 = (tile & 7) * 32;
    const int t = threadIdx.x;

    const size_t base1 = ((size_t)b * NCH2P + c) * NPIX;
    const size_t base2 = ((size_t)b * NCH2P + c + NCH) * NPIX;
    for (int i = t; i < 34 * 34; i += 256) {
        const int hy = i / 34, hx = i % 34;
        const int gy = th0 + hy - 1, gx = tw0 + hx - 1;
        const bool inb = (gy >= 0) & (gy < HWDIM) & (gx >= 0) & (gx < HWDIM);
        const size_t off = (size_t)(inb ? gy : 0) * HWDIM + (inb ? gx : 0);
        s1[hy][hx] = inb ? (float)xin[base1 + off] : 0.f;
        s2[hy][hx] = inb ? (float)xin[base2 + off] : 0.f;
    }
    float k1w[9], k2w[9];
    #pragma unroll
    for (int i = 0; i < 9; ++i) {
        k1w[i] = w_dw[c * 9 + i];
        k2w[i] = w_dw[(c + NCH) * 9 + i];
    }
    __syncthreads();

    float lsum = 0.f;
    const size_t fbase = ((size_t)b * NCH + c) * NPIX;
    #pragma unroll
    for (int ppi = 0; ppi < 4; ++ppi) {
        const int p = t + ppi * 256;
        const int y = p >> 5, xq = p & 31;
        float a1 = 0.f, a2 = 0.f;
        #pragma unroll
        for (int dy = 0; dy < 3; ++dy)
            #pragma unroll
            for (int dx = 0; dx < 3; ++dx) {
                a1 += s1[y + dy][xq + dx] * k1w[dy * 3 + dx];
                a2 += s2[y + dy][xq + dx] * k2w[dy * 3 + dx];
            }
        const float g = 0.5f * a1 * (1.f + erff(a1 * 0.70710678f));  // exact GELU
        const float f = g * a2;
        feat[fbase + (size_t)(th0 + y) * HWDIM + tw0 + xq] = (_Float16)f;
        lsum += f;
    }
    red[t] = lsum;
    __syncthreads();
    for (int s = 128; s > 0; s >>= 1) {
        if (t < s) red[t] += red[t + s];
        __syncthreads();
    }
    if (t == 0) psum[((size_t)b * NCH + c) * 64 + tile] = red[0];
}

// ---------------------------------------------------------------------------
// K3: finish GAP, tiny MLP: mod_b = sigmoid( relu(gap @ w_mod1^T) @ w_mod2^T )
// ---------------------------------------------------------------------------
__global__ __launch_bounds__(256)
void psffn_k3_mod(const float* __restrict__ psum, const float* __restrict__ w_mod1,
                  const float* __restrict__ w_mod2, float* __restrict__ tmod)
{
    __shared__ float sg[NB * NCH];
    __shared__ float sh[NB * 10];
    const int t = threadIdx.x;
    for (int i = t; i < NB * NCH; i += 256) {
        float s = 0.f;
        for (int j = 0; j < 64; ++j) s += psum[(size_t)i * 64 + j];
        sg[i] = s * (1.f / (float)NPIX);
    }
    __syncthreads();
    if (t < NB * 10) {
        const int b = t / 10, j = t % 10;
        float acc = 0.f;
        for (int c = 0; c < NCH; ++c) acc += sg[b * NCH + c] * w_mod1[j * NCH + c];
        sh[t] = fmaxf(acc, 0.f);
    }
    __syncthreads();
    if (t < NB) {
        float m = 0.f;
        for (int j = 0; j < 10; ++j) m += sh[t * 10 + j] * w_mod2[j];
        tmod[t] = 1.f / (1.f + __expf(-m));
    }
}

// ---------------------------------------------------------------------------
// K4a: per-patch spectral transform. For each (b, c, 8x8 patch):
//   spatial = irfft2( rfft2(p) * W_base * (1 + mask*cs*(0.5+mod_b)) )
// Direct 8-point DFTs; all twiddles are multiples of 45 deg -> LDS tables.
// ---------------------------------------------------------------------------
__global__ __launch_bounds__(256)
void psffn_k4a_patch(const _Float16* __restrict__ feat, const float* __restrict__ wbr,
                     const float* __restrict__ wbi, const float* __restrict__ cscale,
                     const float* __restrict__ tmod, _Float16* __restrict__ spat)
{
    __shared__ float sp [32][PP][PP];
    __shared__ float sYr[32][PP][PP];
    __shared__ float sYi[32][PP][PP];
    __shared__ float sFr[32][PP][PFP];
    __shared__ float sFi[32][PP][PFP];
    __shared__ float sHr[32][PP][PFP];
    __shared__ float sHi[32][PP][PFP];
    __shared__ float swr[PP][PFP];
    __shared__ float swi[PP][PFP];
    __shared__ float ct[8], st[8];

    const int b = blockIdx.z, c = blockIdx.y;
    const int t = threadIdx.x, q = t >> 3, r = t & 7;

    if (t < 8) {
        const float ang = 0.78539816339744831f * (float)t;   // 2*pi*t/8
        ct[t] = __cosf(ang);
        st[t] = __sinf(ang);
    }
    if (t < PP * PFP) {
        const int ky = t / PFP, kx = t % PFP;
        const float mask  = __expf(-(float)(ky * ky + kx * kx) * (1.f / 18.f)); // 2*R^2=18
        const float boost = 1.f + mask * (cscale[c] * (0.5f + tmod[b]));
        swr[ky][kx] = wbr[c * (PP * PFP) + t] * boost;
        swi[ky][kx] = wbi[c * (PP * PFP) + t] * boost;
    }

    const int pid = blockIdx.x * 32 + q;
    const int ph = pid >> 5, pw = pid & 31;
    const size_t base = ((size_t)b * NCH + c) * NPIX;
    {
        const _Float16* src = feat + base + (size_t)(ph * 8 + r) * HWDIM + pw * 8;
        #pragma unroll
        for (int xq = 0; xq < 8; ++xq) sp[q][r][xq] = (float)src[xq];
    }
    __syncthreads();

    {   // fft along y: Y[ky][x] = sum_y p[y][x] * e^{-i*2pi*ky*y/8}
        const int ky = r;
        #pragma unroll
        for (int xq = 0; xq < 8; ++xq) {
            float yr = 0.f, yi = 0.f;
            #pragma unroll
            for (int y = 0; y < 8; ++y) {
                const float v = sp[q][y][xq];
                const int m = (ky * y) & 7;
                yr += v * ct[m];
                yi -= v * st[m];
            }
            sYr[q][ky][xq] = yr; sYi[q][ky][xq] = yi;
        }
    }
    __syncthreads();

    {   // rfft along x + complex weight: F[ky][kx] (kx = 0..4)
        const int ky = r;
        #pragma unroll
        for (int kx = 0; kx < PFP; ++kx) {
            float fr = 0.f, fi = 0.f;
            #pragma unroll
            for (int xq = 0; xq < 8; ++xq) {
                const int m = (kx * xq) & 7;
                const float yr = sYr[q][ky][xq], yi = sYi[q][ky][xq];
                fr += yr * ct[m] + yi * st[m];
                fi += yi * ct[m] - yr * st[m];
            }
            const float wr = swr[ky][kx], wi = swi[ky][kx];
            sFr[q][ky][kx] = fr * wr - fi * wi;
            sFi[q][ky][kx] = fr * wi + fi * wr;
        }
    }
    __syncthreads();

    {   // ifft along y: H[y][kx] = (1/8) sum_ky F[ky][kx] * e^{+i*2pi*ky*y/8}
        const int y = r;
        #pragma unroll
        for (int kx = 0; kx < PFP; ++kx) {
            float hr = 0.f, hi = 0.f;
            #pragma unroll
            for (int ky = 0; ky < 8; ++ky) {
                const int m = (ky * y) & 7;
                const float fr = sFr[q][ky][kx], fi = sFi[q][ky][kx];
                hr += fr * ct[m] - fi * st[m];
                hi += fr * st[m] + fi * ct[m];
            }
            sHr[q][y][kx] = hr * 0.125f;
            sHi[q][y][kx] = hi * 0.125f;
        }
    }
    __syncthreads();

    {   // irfft along x (Hermitian extension, real output)
        const int y = r;
        _Float16* dst = spat + base + (size_t)(ph * 8 + y) * HWDIM + pw * 8;
        #pragma unroll
        for (int xq = 0; xq < 8; ++xq) {
            float acc = sHr[q][y][0];
            #pragma unroll
            for (int k = 1; k < 4; ++k) {
                const int m = (k * xq) & 7;
                acc += 2.f * (sHr[q][y][k] * ct[m] - sHi[q][y][k] * st[m]);
            }
            acc += sHr[q][y][4] * ct[(4 * xq) & 7];   // Nyquist bin (+-1)
            dst[xq] = (_Float16)(acc * 0.125f);
        }
    }
}

// ---------------------------------------------------------------------------
// K4b: out-projection GEMM  out[b,o,n] = sum_c w_out[o,c] * spatial[b,c,n]
//     M = 64 (4 tiles), K = 170 -> 192 (6 k-steps of 32, zero padded at
//     staging), N tile = 64. w_out and spatial both staged in LDS as f16.
// ---------------------------------------------------------------------------
__global__ __launch_bounds__(256)
void psffn_k4b_outproj(const _Float16* __restrict__ spat, const float* __restrict__ w_out,
                       float* __restrict__ out)
{
    __shared__ __align__(16) _Float16 sw2[64 * 192]; // [o][k], k >= 170 zeroed
    __shared__ __align__(16) _Float16 sb[64][200];   // [n][k], 192 padded to 200
    const int b  = blockIdx.y;
    const int n0 = blockIdx.x * 64;
    const int t  = threadIdx.x;

    for (int i = t; i < 64 * 192; i += 256) {
        const int o = i / 192, k = i % 192;
        sw2[i] = (k < NCH) ? (_Float16)w_out[o * NCH + k] : (_Float16)0.f;
    }
    for (int i = t; i < 192 * 64; i += 256) {
        const int k = i >> 6, n = i & 63;
        sb[n][k] = (k < NCH) ? spat[((size_t)b * NCH + k) * NPIX + n0 + n]
                             : (_Float16)0.f;
    }
    __syncthreads();

    const int wave = t >> 5, lane = t & 31;
    const int nsub = lane & 15, hh = lane >> 4;
    const int mt = wave & 3, ngrp = wave >> 2;

    const v8f vzero = {0.f, 0.f, 0.f, 0.f, 0.f, 0.f, 0.f, 0.f};
    v8f acc[2];
    acc[0] = vzero; acc[1] = vzero;

    #pragma unroll
    for (int ks = 0; ks < 6; ++ks) {
        const _Float16* pa = &sw2[(mt * 16 + nsub) * 192 + ks * 32 + hh * 8];
        const v16h a = cat8(*(const v8h*)pa, *(const v8h*)(pa + 16));
        #pragma unroll
        for (int j = 0; j < 2; ++j) {
            const int nt = ngrp * 2 + j;
            const _Float16* pb = &sb[nt * 16 + nsub][ks * 32 + hh * 8];
            const v16h bf = cat8(*(const v8h*)pb, *(const v8h*)(pb + 16));
            acc[j] = __builtin_amdgcn_wmma_f32_16x16x32_f16(
                         false, a, false, bf, (short)0, acc[j], false, false);
        }
    }
    #pragma unroll
    for (int j = 0; j < 2; ++j) {
        const int nt = ngrp * 2 + j;
        #pragma unroll
        for (int v = 0; v < 8; ++v) {
            const int o = mt * 16 + v + 8 * hh;
            out[((size_t)b * NDIM + o) * NPIX + n0 + nt * 16 + nsub] = acc[j][v];
        }
    }
}

// ---------------------------------------------------------------------------
extern "C" void kernel_launch(void* const* d_in, const int* in_sizes, int n_in,
                              void* d_out, int out_size, void* d_ws, size_t ws_size,
                              hipStream_t stream)
{
    (void)in_sizes; (void)n_in; (void)out_size; (void)ws_size;

    const float* x      = (const float*)d_in[0];   // (4,64,256,256)
    const float* w_in   = (const float*)d_in[1];   // (340,64)
    const float* w_dw   = (const float*)d_in[2];   // (340,1,3,3)
    const float* w_out  = (const float*)d_in[3];   // (64,170)
    const float* wbr    = (const float*)d_in[4];   // (170,1,1,8,5)
    const float* wbi    = (const float*)d_in[5];   // (170,1,1,8,5)
    const float* cs     = (const float*)d_in[6];   // (170,)
    const float* w_mod1 = (const float*)d_in[7];   // (10,170)
    const float* w_mod2 = (const float*)d_in[8];   // (1,10)
    float* out = (float*)d_out;                    // (4,64,256,256) fp32

    // workspace layout (fp16 intermediates to halve HBM traffic)
    _Float16* xin  = (_Float16*)d_ws;                              // 4*352*65536 halves (padded)
    _Float16* feat = xin + (size_t)NB * NCH2P * NPIX;              // 4*170*65536 halves
    float*    psum = (float*)(feat + (size_t)NB * NCH * NPIX);     // 4*170*64 floats
    float*    tmod = psum + (size_t)NB * NCH * 64;                 // 4 floats
    _Float16* spat = xin;   // x_in is dead after K2 -> reuse for spatial

    psffn_k1_inproj  <<<dim3(NPIX / 128, NB),      256, 0, stream>>>(x, w_in, xin);
    psffn_k2_dwgate  <<<dim3(64, NCH, NB),         256, 0, stream>>>(xin, w_dw, feat, psum);
    psffn_k3_mod     <<<dim3(1),                   256, 0, stream>>>(psum, w_mod1, w_mod2, tmod);
    psffn_k4a_patch  <<<dim3(32, NCH, NB),         256, 0, stream>>>(feat, wbr, wbi, cs, tmod, spat);
    psffn_k4b_outproj<<<dim3(NPIX / 64, NB),       256, 0, stream>>>(spat, w_out, out);
}
